// MoEBlock_32581621907469
// MI455X (gfx1250) — compile-verified, matrix operand order checked by reading, not run on hardware
//
#include <hip/hip_runtime.h>
#include <hip/hip_fp16.h>

// ---------------------------------------------------------------------------
// Types
// ---------------------------------------------------------------------------
typedef _Float16 h16;
typedef __attribute__((ext_vector_type(16))) _Float16 v16h;
typedef __attribute__((ext_vector_type(8)))  _Float16 v8h;
typedef __attribute__((ext_vector_type(4)))  _Float16 v4h;
typedef __attribute__((ext_vector_type(8)))  float    v8f;
typedef __attribute__((ext_vector_type(4)))  unsigned int u32x4;
typedef __attribute__((ext_vector_type(8)))  int i32x8;
typedef __attribute__((ext_vector_type(4)))  int i32x4;

#define WMMA_F16(a, b, c) \
  __builtin_amdgcn_wmma_f32_16x16x32_f16(false, (a), false, (b), (short)0, (c), false, false)

// Problem dims (fixed by reference)
constexpr int Bb = 4, Tt = 1024, Cc = 1024, Hh = 16, Nn = 64, Ff = 1024, Ee = 16;
constexpr int NT = Bb * Tt;                   // 4096 tokens
constexpr float EPSR = 1.1920929e-07f;

// TDM availability
#ifndef CDNA5_USE_TDM
#if defined(__has_builtin)
#if __has_builtin(__builtin_amdgcn_tensor_load_to_lds)
#define CDNA5_USE_TDM 1
#else
#define CDNA5_USE_TDM 0
#endif
#else
#define CDNA5_USE_TDM 0
#endif
#endif

// ---------------------------------------------------------------------------
// Fragment-layout helpers (wave32 WMMA 16x16x32 f16)
// A frag element (row m in 0..15, k in 0..31):
//   lane = (m&15) + 16*((k>>3)&1) ; e = ((k>>4)<<3) + (k&7)
// B frag mirrors with column index in place of m. One fragment = lane*16 halfs
// contiguous, so a lane reads its fragment as a single 32B LDS read.
// ---------------------------------------------------------------------------
__device__ __forceinline__ int frag_off(int rc, int k) {
  int lane = (rc & 15) + (((k >> 3) & 1) << 4);
  int e = ((k >> 4) << 3) + (k & 7);
  return lane * 16 + e;
}

__device__ __forceinline__ float redmax16(float v) {
#pragma unroll
  for (int o = 8; o; o >>= 1) v = fmaxf(v, __shfl_xor(v, o, 32));
  return v;
}
__device__ __forceinline__ float redsum16(float v) {
#pragma unroll
  for (int o = 8; o; o >>= 1) v += __shfl_xor(v, o, 32);
  return v;
}
__device__ __forceinline__ float redsum32(float v) {
#pragma unroll
  for (int o = 16; o; o >>= 1) v += __shfl_xor(v, o, 32);
  return v;
}
__device__ __forceinline__ float atomAddF(float* p, float v) {
  return __hip_atomic_fetch_add(p, v, __ATOMIC_RELAXED, __HIP_MEMORY_SCOPE_AGENT);
}
__device__ __forceinline__ int atomAddI(int* p, int v) {
  return __hip_atomic_fetch_add(p, v, __ATOMIC_RELAXED, __HIP_MEMORY_SCOPE_AGENT);
}

// Cooperative swizzled tile stagers (128 threads)
// A tile: 64 rows x 32 k, row-major f16 source with leading dim ldk.
__device__ __forceinline__ void stage_a_tile(const h16* __restrict__ src_base, int ldk,
                                             h16* __restrict__ dst, int tid) {
  int mm = tid >> 1;                // 0..63
  int kk = (tid & 1) * 16;          // 0 or 16
  const h16* s = src_base + (size_t)mm * ldk + kk;
  v8h x0 = *(const v8h*)s;
  v8h x1 = *(const v8h*)(s + 8);
  h16* d = dst + (mm >> 4) * 512;
  int ml = mm & 15;
#pragma unroll
  for (int i = 0; i < 8; ++i) {
    d[frag_off(ml, kk + i)] = x0[i];
    d[frag_off(ml, kk + 8 + i)] = x1[i];
  }
}
// B tile: 32 k x 64 n, row-major f16 source with leading dim ldn.
__device__ __forceinline__ void stage_b_tile(const h16* __restrict__ src_base, int ldn,
                                             h16* __restrict__ dst, int tid) {
  int kk = tid >> 2;                // 0..31
  int nn = (tid & 3) * 16;          // 0,16,32,48
  const h16* s = src_base + (size_t)kk * ldn + nn;
  v8h x0 = *(const v8h*)s;
  v8h x1 = *(const v8h*)(s + 8);
  h16* d = dst + (nn >> 4) * 512;
#pragma unroll
  for (int i = 0; i < 8; ++i) {
    d[frag_off(i, kk)] = x0[i];
    d[frag_off(8 + i, kk)] = x1[i];
  }
}
// Linear 32x64 tile stager (for TDM-compatible layout)
__device__ __forceinline__ void stage_lin(h16* __restrict__ dst, const h16* __restrict__ src,
                                          int ld, int tid) {
  int r = tid >> 2, c0 = (tid & 3) * 16;
  const h16* s = src + (size_t)r * ld + c0;
  *(v8h*)(dst + r * 64 + c0) = *(const v8h*)s;
  *(v8h*)(dst + r * 64 + c0 + 8) = *(const v8h*)(s + 8);
}
// Build a B fragment (ntile j) straight from a linear 32x64 LDS tile.
__device__ __forceinline__ v16h bfrag_linear(const h16* __restrict__ sW, int j, int lane) {
  int n = j * 16 + (lane & 15), hf = lane >> 4;
  v16h f;
#pragma unroll
  for (int g = 0; g < 2; ++g)
#pragma unroll
    for (int i = 0; i < 8; ++i)
      f[g * 8 + i] = sW[(g * 16 + hf * 8 + i) * 64 + n];
  return f;
}

#if CDNA5_USE_TDM
// Tensor Data Mover: async 2D f16 tile (rows x cols, row stride in elements)
// from global into LDS. Descriptor per CDNA5 ISA D# spec (CDNA5 ISA).
// 6-arg builtin form: (g0 u32x4, g1 i32x8, g2 i32x4, g3 i32x4, g4 i32x8, cpol).
__device__ __forceinline__ void tdm_load_2d(void* ldsptr, const void* gptr,
                                            unsigned rows, unsigned cols,
                                            unsigned long long stride_elems) {
  unsigned long long ga = (unsigned long long)(uintptr_t)gptr;
  unsigned lds = (unsigned)(uintptr_t)ldsptr;
  u32x4 g0;
  g0[0] = 1u;                                          // count=1, user mode
  g0[1] = lds;                                         // lds_addr
  g0[2] = (unsigned)(ga & 0xffffffffull);              // global_addr lo
  g0[3] = (unsigned)((ga >> 32) & 0x01ffffffull) | (2u << 30); // addr hi | type=2
  i32x8 g1;
  g1[0] = (int)(1u << 16);                             // wg_mask=0, data_size=1 (2B)
  g1[1] = (int)((cols & 0xffffu) << 16);               // tensor_dim0[15:0]
  g1[2] = (int)(((cols >> 16) & 0xffffu) | ((rows & 0xffffu) << 16)); // dim0 hi | dim1 lo
  g1[3] = (int)(((rows >> 16) & 0xffffu) | ((cols & 0xffffu) << 16)); // dim1 hi | tile_dim0
  g1[4] = (int)(rows & 0xffffu);                       // tile_dim1, tile_dim2=0
  g1[5] = (int)(stride_elems & 0xffffffffull);         // tensor_dim0_stride lo
  g1[6] = (int)((stride_elems >> 32) & 0xffffull);     // stride hi | dim1_stride lo=0
  g1[7] = 0;
  i32x4 g2; g2[0] = g2[1] = g2[2] = g2[3] = 0;
  i32x4 g3; g3[0] = g3[1] = g3[2] = g3[3] = 0;
  i32x8 g4;
#pragma unroll
  for (int i = 0; i < 8; ++i) g4[i] = 0;
  __builtin_amdgcn_tensor_load_to_lds(g0, g1, g2, g3, g4, 0);
}
#endif

// ---------------------------------------------------------------------------
// Small utility kernels
// ---------------------------------------------------------------------------
__global__ __launch_bounds__(256) void k_init(float* moe, float* msc, float* moh, int* cnt,
                                              size_t n) {
  size_t i = (size_t)blockIdx.x * 256 + threadIdx.x;
  size_t st = (size_t)gridDim.x * 256;
  for (size_t j = i; j < n; j += st) moe[j] = 0.f;
  if (i < 16) { msc[i] = 0.f; moh[i] = 0.f; cnt[i] = 0; }
}

__global__ __launch_bounds__(256) void k_convert(const float* __restrict__ s,
                                                 h16* __restrict__ d, size_t n) {
  size_t i = (size_t)blockIdx.x * 256 + threadIdx.x;
  size_t st = (size_t)gridDim.x * 256;
  for (; i < n; i += st) d[i] = (h16)s[i];
}

template <typename Tin>
__global__ __launch_bounds__(256) void k_rmsnorm(const Tin* __restrict__ X,
                                                 h16* __restrict__ Y) {
  __shared__ float red[8];
  __shared__ float stot;
  const int row = blockIdx.x, tid = threadIdx.x;
  const Tin* xr = X + (size_t)row * Cc;
  float ss = 0.f;
  for (int c = tid; c < Cc; c += 256) { float v = (float)xr[c]; ss += v * v; }
  ss = redsum32(ss);
  if ((tid & 31) == 0) red[tid >> 5] = ss;
  __syncthreads();
  if (tid == 0) {
    float t = 0.f;
#pragma unroll
    for (int i = 0; i < 8; ++i) t += red[i];
    stot = rsqrtf(t * (1.0f / Cc) + EPSR);
  }
  __syncthreads();
  float sc = stot;
  for (int c = tid; c < Cc; c += 256) Y[(size_t)row * Cc + c] = (h16)((float)xr[c] * sc);
}

// per-(token,head) rms + rope, in place on f16 buffer; grid (NT, 2), 256 thr
__global__ __launch_bounds__(256) void k_rope(h16* __restrict__ X,
                                              const float* __restrict__ Cos,
                                              const float* __restrict__ Sin) {
  const int tid = threadIdx.x, wave = tid >> 5, lane = tid & 31;
  const int row = blockIdx.x;
  const int h = blockIdx.y * 8 + wave;
  const int t = row & (Tt - 1);
  h16* p = X + (size_t)row * (Hh * Nn) + h * Nn;
  float a = (float)p[lane], b = (float)p[lane + 32];
  float ss = a * a + b * b;
  ss = redsum32(ss);
  float sc = rsqrtf(ss * (1.0f / Nn) + EPSR);
  a *= sc; b *= sc;
  float c = Cos[t * 32 + lane], s = Sin[t * 32 + lane];
  p[lane] = (h16)(a * c - b * s);
  p[lane + 32] = (h16)(a * s + b * c);
}

__global__ __launch_bounds__(256) void k_og(const h16* __restrict__ O,
                                            const h16* __restrict__ G,
                                            h16* __restrict__ OG, size_t n) {
  size_t i = (size_t)blockIdx.x * 256 + threadIdx.x;
  size_t st = (size_t)gridDim.x * 256;
  for (; i < n; i += st) {
    float g = (float)G[i];
    float sg = 1.f / (1.f + __expf(-g));
    OG[i] = (h16)((float)O[i] * sg);
  }
}

__global__ __launch_bounds__(256) void k_combine(const float* __restrict__ xa,
                                                 const float* __restrict__ moe,
                                                 const float* __restrict__ sg,
                                                 const float* __restrict__ svbuf,
                                                 const float* __restrict__ svbias,
                                                 float* __restrict__ out, size_t n) {
  size_t i = (size_t)blockIdx.x * 256 + threadIdx.x;
  size_t st = (size_t)gridDim.x * 256;
  for (; i < n; i += st) {
    size_t t = i >> 10;
    int c = (int)(i & 1023);
    out[i] = xa[i] + moe[i] + sg[t] * (svbuf[i] + svbias[c]);
  }
}

__global__ void k_aux(const float* __restrict__ msc, const float* __restrict__ moh,
                      float* __restrict__ out) {
  if (threadIdx.x == 0 && blockIdx.x == 0) {
    float s = 0.f;
#pragma unroll
    for (int e = 0; e < Ee; ++e) s += msc[e] * moh[e];
    out[0] = (float)Ee * s * 0.001f;
  }
}

// one wave per token: sg[t] = sigmoid(xt . shared_gate_w)
__global__ __launch_bounds__(128) void k_sg(const h16* __restrict__ Xt,
                                            const float* __restrict__ w,
                                            float* __restrict__ sg) {
  int wave = threadIdx.x >> 5, lane = threadIdx.x & 31;
  int t = blockIdx.x * 4 + wave;
  const h16* xr = Xt + (size_t)t * Cc;
  float a = 0.f;
  for (int c = lane; c < Cc; c += 32) a += (float)xr[c] * w[c];
  a = redsum32(a);
  if (lane == 0) sg[t] = 1.f / (1.f + __expf(-a));
}

// ---------------------------------------------------------------------------
// Generic 64x64-tile WMMA GEMM: D = epilogue(A[f16 MxK] @ B[f16 KxN])
// EPI: 0=f16 store, 1=f32 store + residual, 2=f16 sq(relu(x+bias)), 3=f32 store
// ---------------------------------------------------------------------------
template <int EPI>
__global__ __launch_bounds__(128) void k_gemm(const h16* __restrict__ A,
                                              const h16* __restrict__ Bw,
                                              const float* __restrict__ resid,
                                              const float* __restrict__ bias,
                                              float* __restrict__ Df,
                                              h16* __restrict__ Dh, int M, int N, int K) {
  __shared__ h16 sA[2][2048];
  __shared__ h16 sB[2][2048];
  const int tid = threadIdx.x, wave = tid >> 5, lane = tid & 31;
  const int m0 = blockIdx.x * 64, n0 = blockIdx.y * 64;
  v8f acc[4];
#pragma unroll
  for (int j = 0; j < 4; ++j)
#pragma unroll
    for (int r = 0; r < 8; ++r) acc[j][r] = 0.f;
  const int nsteps = K >> 5;

  stage_a_tile(A + (size_t)m0 * K, K, sA[0], tid);
  stage_b_tile(Bw + n0, N, sB[0], tid);
  __syncthreads();
  for (int ks = 0; ks < nsteps; ++ks) {
    const int cur = ks & 1, nxt = cur ^ 1;
    if (ks + 1 < nsteps) {
      stage_a_tile(A + (size_t)m0 * K + (ks + 1) * 32, K, sA[nxt], tid);
      stage_b_tile(Bw + (size_t)(ks + 1) * 32 * N + n0, N, sB[nxt], tid);
    }
    if (ks + 2 < nsteps)
      __builtin_prefetch((const void*)(Bw + (size_t)(ks + 2) * 32 * N + n0), 0, 1);
    v16h af = *(const v16h*)&sA[cur][wave * 512 + lane * 16];
#pragma unroll
    for (int j = 0; j < 4; ++j) {
      v16h bf = *(const v16h*)&sB[cur][j * 512 + lane * 16];
      acc[j] = WMMA_F16(af, bf, acc[j]);
    }
    __syncthreads();
  }
  const int hf = lane >> 4, nl = lane & 15;
#pragma unroll
  for (int j = 0; j < 4; ++j)
#pragma unroll
    for (int r = 0; r < 8; ++r) {
      int m = m0 + wave * 16 + r + 8 * hf;
      int n = n0 + j * 16 + nl;
      size_t idx = (size_t)m * N + n;
      float v = acc[j][r];
      if (EPI == 0) {
        Dh[idx] = (h16)v;
      } else if (EPI == 1) {
        Df[idx] = v + resid[idx];
      } else if (EPI == 2) {
        v += bias[n];
        v = v > 0.f ? v * v : 0.f;
        Dh[idx] = (h16)v;
      } else {
        Df[idx] = v;
      }
    }
}

// ---------------------------------------------------------------------------
// Flash attention, causal.  grid (T/64, B*H), 128 threads (4 waves).
// Wave w owns query rows [qt*64 + w*16, +16).  Key chunks of 64.
// ---------------------------------------------------------------------------
__global__ __launch_bounds__(128) void k_attn(const h16* __restrict__ Q,
                                              const h16* __restrict__ Km,
                                              const h16* __restrict__ V,
                                              h16* __restrict__ O) {
  __shared__ h16 sK[4096];   // [kstep(2)][ntile(4)][512]
  __shared__ h16 sV[4096];
  __shared__ h16 sP[4096];   // per-wave 1024
  const int tid = threadIdx.x, wave = tid >> 5, lane = tid & 31;
  const int qt = blockIdx.x;
  const int b = blockIdx.y >> 4, h = blockIdx.y & 15;
  const int ld = Hh * Nn;
  const size_t base = (size_t)b * Tt * ld + h * Nn;
  const int mloc = lane & 15, hf = lane >> 4;

  // Q fragments for this wave's 16 rows, pre-scaled by 1/sqrt(N)
  v16h aq[2];
  {
    const h16* qr = Q + base + (size_t)(qt * 64 + wave * 16 + mloc) * ld;
#pragma unroll
    for (int ks = 0; ks < 2; ++ks) {
      v8h lo = *(const v8h*)(qr + ks * 32 + hf * 8);
      v8h hi = *(const v8h*)(qr + ks * 32 + 16 + hf * 8);
      v16h f;
#pragma unroll
      for (int i = 0; i < 8; ++i) {
        f[i] = (h16)((float)lo[i] * 0.125f);
        f[8 + i] = (h16)((float)hi[i] * 0.125f);
      }
      aq[ks] = f;
    }
  }
  float rowm[8], rows[8];
#pragma unroll
  for (int r = 0; r < 8; ++r) { rowm[r] = -3.0e38f; rows[r] = 0.f; }
  v8f oacc[4];
#pragma unroll
  for (int j = 0; j < 4; ++j)
#pragma unroll
    for (int r = 0; r < 8; ++r) oacc[j][r] = 0.f;

  for (int scn = 0; scn <= qt; ++scn) {
    { // stage K chunk (B-frag: n=key, k=head dim) and V chunk (n=hd, k=key)
      int key = tid >> 1, c0 = (tid & 1) * 16;
      const h16* srcK = Km + base + (size_t)(scn * 64 + key) * ld;
      int kl = key & 15;
#pragma unroll
      for (int ks = 0; ks < 2; ++ks) {
        v8h x0 = *(const v8h*)(srcK + ks * 32 + c0);
        v8h x1 = *(const v8h*)(srcK + ks * 32 + c0 + 8);
        h16* dst = sK + ks * 2048 + (key >> 4) * 512;
#pragma unroll
        for (int i = 0; i < 8; ++i) {
          dst[frag_off(kl, c0 + i)] = x0[i];
          dst[frag_off(kl, c0 + 8 + i)] = x1[i];
        }
      }
      const h16* srcV = V + base + (size_t)(scn * 64 + key) * ld;
#pragma unroll
      for (int p = 0; p < 2; ++p) {
        int c0v = p * 32 + (tid & 1) * 16;
        v8h x0 = *(const v8h*)(srcV + c0v);
        v8h x1 = *(const v8h*)(srcV + c0v + 8);
        h16* dst = sV + (key >> 5) * 2048 + (c0v >> 4) * 512;
        int kk = key & 31;
#pragma unroll
        for (int i = 0; i < 8; ++i) {
          dst[frag_off(i, kk)] = x0[i];
          dst[frag_off(8 + i, kk)] = x1[i];
        }
      }
    }
    __syncthreads();

    v8f sacc[4];
#pragma unroll
    for (int j = 0; j < 4; ++j)
#pragma unroll
      for (int r = 0; r < 8; ++r) sacc[j][r] = 0.f;
#pragma unroll
    for (int j = 0; j < 4; ++j)
#pragma unroll
      for (int ks = 0; ks < 2; ++ks) {
        v16h bf = *(const v16h*)&sK[ks * 2048 + j * 512 + lane * 16];
        sacc[j] = WMMA_F16(aq[ks], bf, sacc[j]);
      }

    // online softmax update + write P to per-wave A-fragment scratch
    h16* sPw = sP + wave * 1024;
#pragma unroll
    for (int r = 0; r < 8; ++r) {
      int mg = qt * 64 + wave * 16 + r + 8 * hf;
      float mx = -3.0e38f;
#pragma unroll
      for (int j = 0; j < 4; ++j) {
        int sgk = scn * 64 + j * 16 + mloc;
        if (sgk > mg) sacc[j][r] = -3.0e38f;
        mx = fmaxf(mx, sacc[j][r]);
      }
      mx = redmax16(mx);
      float nm = fmaxf(rowm[r], mx);
      float corr = __expf(rowm[r] - nm);
      rowm[r] = nm;
      float ls = 0.f;
#pragma unroll
      for (int j = 0; j < 4; ++j) {
        float p = __expf(sacc[j][r] - nm);
        ls += p;
        int kl = j * 16 + mloc;
        sPw[(kl >> 5) * 512 + frag_off(r + 8 * hf, kl & 31)] = (h16)p;
      }
      ls = redsum16(ls);
      rows[r] = rows[r] * corr + ls;
#pragma unroll
      for (int j = 0; j < 4; ++j) oacc[j][r] *= corr;
    }
    // O += P @ V  (intra-wave LDS write->read, DS ops are in order per wave)
#pragma unroll
    for (int ks = 0; ks < 2; ++ks) {
      v16h ap = *(const v16h*)&sPw[ks * 512 + lane * 16];
#pragma unroll
      for (int j = 0; j < 4; ++j) {
        v16h bv = *(const v16h*)&sV[ks * 2048 + j * 512 + lane * 16];
        oacc[j] = WMMA_F16(ap, bv, oacc[j]);
      }
    }
    __syncthreads();
  }
#pragma unroll
  for (int r = 0; r < 8; ++r) {
    float inv = 1.f / rows[r];
    int row = qt * 64 + wave * 16 + r + 8 * hf;
    h16* orow = O + base + (size_t)row * ld;
#pragma unroll
    for (int j = 0; j < 4; ++j) orow[j * 16 + mloc] = (h16)(oacc[j][r] * inv);
  }
}

// ---------------------------------------------------------------------------
// Router: one wave per token; softmax over E=16, top-2, per-expert lists,
// aux-loss accumulators.
// ---------------------------------------------------------------------------
__global__ __launch_bounds__(128) void k_router(const h16* __restrict__ Xt,
                                                const float* __restrict__ Gw,
                                                int* __restrict__ ti, float* __restrict__ tw,
                                                float* __restrict__ msc,
                                                float* __restrict__ moh,
                                                int* __restrict__ cnt,
                                                int* __restrict__ lists) {
  int wave = threadIdx.x >> 5, lane = threadIdx.x & 31;
  int t = blockIdx.x * 4 + wave;
  const h16* xr = Xt + (size_t)t * Cc;
  float acc[16];
#pragma unroll
  for (int e = 0; e < 16; ++e) acc[e] = 0.f;
  for (int c = lane; c < Cc; c += 32) {
    float xv = (float)xr[c];
    const float* g = Gw + c * 16;
#pragma unroll
    for (int e = 0; e < 16; ++e) acc[e] += xv * g[e];
  }
#pragma unroll
  for (int e = 0; e < 16; ++e) acc[e] = redsum32(acc[e]);
  if (lane == 0) {
    float mx = acc[0];
#pragma unroll
    for (int e = 1; e < 16; ++e) mx = fmaxf(mx, acc[e]);
    float p[16], s = 0.f;
#pragma unroll
    for (int e = 0; e < 16; ++e) { p[e] = __expf(acc[e] - mx); s += p[e]; }
    float inv = 1.f / s;
#pragma unroll
    for (int e = 0; e < 16; ++e) p[e] *= inv;
    int i0 = 0;
#pragma unroll
    for (int e = 1; e < 16; ++e) if (p[e] > p[i0]) i0 = e;
    int i1 = (i0 == 0) ? 1 : 0;
#pragma unroll
    for (int e = 0; e < 16; ++e) if (e != i0 && p[e] > p[i1]) i1 = e;
    float wsum = p[i0] + p[i1] + 1e-6f;
    ti[t * 2] = i0; ti[t * 2 + 1] = i1;
    tw[t * 2] = p[i0] / wsum; tw[t * 2 + 1] = p[i1] / wsum;
#pragma unroll
    for (int e = 0; e < 16; ++e) atomAddF(&msc[e], p[e] * (1.f / NT));
    atomAddF(&moh[i0], 0.5f / NT);
    atomAddF(&moh[i1], 0.5f / NT);
    int pos = atomAddI(&cnt[i0], 1);
    lists[i0 * NT + pos] = t * 2;
    pos = atomAddI(&cnt[i1], 1);
    lists[i1 * NT + pos] = t * 2 + 1;
  }
}

// ---------------------------------------------------------------------------
// Per-expert tile: 16 routed tokens through expert e.
// grid (NT/16, E), 128 threads.  xt and h live in LDS in A-frag layout;
// W1/W2 tiles streamed via the Tensor Data Mover (async, TENSORcnt).
// ---------------------------------------------------------------------------
__global__ __launch_bounds__(128) void k_moe(const h16* __restrict__ Xt,
                                             const h16* __restrict__ W1h,
                                             const h16* __restrict__ W2h,
                                             const int* __restrict__ cnt,
                                             const int* __restrict__ lists,
                                             const float* __restrict__ tw,
                                             float* __restrict__ Out) {
  const int e = blockIdx.y, tile = blockIdx.x;
  const int ne = cnt[e];
  if (tile * 16 >= ne) return;
  __shared__ int toks[16];
  __shared__ float wts[16];
  __shared__ h16 sA[32 * 512];      // xt rows, A-frag layout, full K=1024
  __shared__ h16 sH[32 * 512];      // h, A-frag layout, full K=1024
  __shared__ h16 sW[2][2048];       // linear 32x64 weight tiles
  const int tid = threadIdx.x, wave = tid >> 5, lane = tid & 31;
  const int mloc = lane & 15, hf = lane >> 4;

  if (tid < 16) {
    int idx = tile * 16 + tid;
    if (idx < ne) {
      int pk = lists[e * NT + idx];
      toks[tid] = pk >> 1;
      wts[tid] = tw[pk];
    } else {
      toks[tid] = -1;
      wts[tid] = 0.f;
    }
  }
  __syncthreads();
  { // gather 16 xt rows into fragment layout
    int m = tid >> 3;
    int tk = toks[m] < 0 ? 0 : toks[m];
    const h16* src = Xt + (size_t)tk * Cc;
    int k0 = (tid & 7) * 4;
    for (int kt = 0; kt < 32; ++kt) {
      v4h x = *(const v4h*)(src + kt * 32 + k0);
      h16* d = &sA[kt * 512];
#pragma unroll
      for (int i = 0; i < 4; ++i) d[frag_off(m, k0 + i)] = x[i];
    }
  }
  __syncthreads();

  // ---------------- phase 1: h = sq(relu(xt @ W1[e])) ----------------
  const h16* W1e = W1h + (size_t)e * Cc * Ff;
  for (int nb = 0; nb < Ff / 64; ++nb) {
    v8f acc;
#pragma unroll
    for (int r = 0; r < 8; ++r) acc[r] = 0.f;
    const h16* bsrc = W1e + nb * 64;
#if CDNA5_USE_TDM
    if (wave == 0) tdm_load_2d(&sW[0][0], bsrc, 32, 64, (unsigned long long)Ff);
    for (int ks = 0; ks < 32; ++ks) {
      const int cur = ks & 1;
      __syncthreads();  // prior readers of sW[cur^1] done
      if (wave == 0) {
        if (ks + 1 < 32) {
          tdm_load_2d(&sW[cur ^ 1][0], bsrc + (size_t)(ks + 1) * 32 * Ff, 32, 64,
                      (unsigned long long)Ff);
          __builtin_amdgcn_s_wait_tensorcnt((short)1);
        } else {
          __builtin_amdgcn_s_wait_tensorcnt((short)0);
        }
      }
      __syncthreads();  // sW[cur] complete for everyone
      v16h af = *(const v16h*)&sA[ks * 512 + lane * 16];
      v16h bf = bfrag_linear(sW[cur], wave, lane);
      acc = WMMA_F16(af, bf, acc);
    }
#else
    stage_lin(sW[0], bsrc, Ff, tid);
    __syncthreads();
    for (int ks = 0; ks < 32; ++ks) {
      const int cur = ks & 1;
      if (ks + 1 < 32) stage_lin(sW[cur ^ 1], bsrc + (size_t)(ks + 1) * 32 * Ff, Ff, tid);
      v16h af = *(const v16h*)&sA[ks * 512 + lane * 16];
      v16h bf = bfrag_linear(sW[cur], wave, lane);
      acc = WMMA_F16(af, bf, acc);
      __syncthreads();
    }
#endif
#pragma unroll
    for (int r = 0; r < 8; ++r) {
      float v = acc[r];
      v = v > 0.f ? v * v : 0.f;
      int fc = nb * 64 + wave * 16 + mloc;
      sH[(fc >> 5) * 512 + frag_off(r + 8 * hf, fc & 31)] = (h16)v;
    }
  }
  __syncthreads();

  // ---------------- phase 2: y = h @ W2[e], scatter-add ----------------
  const h16* W2e = W2h + (size_t)e * Ff * Cc;
  for (int nb = 0; nb < Cc / 64; ++nb) {
    v8f acc;
#pragma unroll
    for (int r = 0; r < 8; ++r) acc[r] = 0.f;
    const h16* bsrc = W2e + nb * 64;
#if CDNA5_USE_TDM
    if (wave == 0) tdm_load_2d(&sW[0][0], bsrc, 32, 64, (unsigned long long)Cc);
    for (int ks = 0; ks < 32; ++ks) {
      const int cur = ks & 1;
      __syncthreads();
      if (wave == 0) {
        if (ks + 1 < 32) {
          tdm_load_2d(&sW[cur ^ 1][0], bsrc + (size_t)(ks + 1) * 32 * Cc, 32, 64,
                      (unsigned long long)Cc);
          __builtin_amdgcn_s_wait_tensorcnt((short)1);
        } else {
          __builtin_amdgcn_s_wait_tensorcnt((short)0);
        }
      }
      __syncthreads();
      v16h af = *(const v16h*)&sH[ks * 512 + lane * 16];
      v16h bf = bfrag_linear(sW[cur], wave, lane);
      acc = WMMA_F16(af, bf, acc);
    }
#else
    stage_lin(sW[0], bsrc, Cc, tid);
    __syncthreads();
    for (int ks = 0; ks < 32; ++ks) {
      const int cur = ks & 1;
      if (ks + 1 < 32) stage_lin(sW[cur ^ 1], bsrc + (size_t)(ks + 1) * 32 * Cc, Cc, tid);
      v16h af = *(const v16h*)&sH[ks * 512 + lane * 16];
      v16h bf = bfrag_linear(sW[cur], wave, lane);
      acc = WMMA_F16(af, bf, acc);
      __syncthreads();
    }
#endif
#pragma unroll
    for (int r = 0; r < 8; ++r) {
      int mrow = r + 8 * hf;
      int tk = toks[mrow];
      if (tk >= 0)
        atomAddF(&Out[(size_t)tk * Cc + nb * 64 + wave * 16 + mloc], wts[mrow] * acc[r]);
    }
  }
}

// ---------------------------------------------------------------------------
// Host launch
// ---------------------------------------------------------------------------
extern "C" void kernel_launch(void* const* d_in, const int* in_sizes, int n_in, void* d_out,
                              int out_size, void* d_ws, size_t ws_size, hipStream_t stream) {
  const float* x    = (const float*)d_in[0];
  const float* cosb = (const float*)d_in[1];
  const float* sinb = (const float*)d_in[2];
  const float* Wq   = (const float*)d_in[3];
  const float* Wk   = (const float*)d_in[4];
  const float* Wv   = (const float*)d_in[5];
  const float* Wg   = (const float*)d_in[6];
  const float* Wo   = (const float*)d_in[7];
  const float* gate_w = (const float*)d_in[8];
  const float* W1   = (const float*)d_in[9];
  const float* W2   = (const float*)d_in[10];
  const float* sgw  = (const float*)d_in[11];
  const float* skw  = (const float*)d_in[12];
  const float* skb  = (const float*)d_in[13];
  const float* svw  = (const float*)d_in[14];
  const float* svb  = (const float*)d_in[15];
  (void)in_sizes; (void)n_in; (void)out_size; (void)ws_size;

  const size_t P = (size_t)NT * Cc;           // 4M activations
  const size_t W = (size_t)Cc * (Hh * Nn);    // 1M per proj weight
  char* ws = (char*)d_ws;
  size_t off = 0;
  auto alloc = [&](size_t bytes) -> void* {
    void* p = ws + off;
    off = (off + bytes + 255) & ~(size_t)255;
    return p;
  };
  h16* xn    = (h16*)alloc(P * 2);
  h16* qh    = (h16*)alloc(P * 2);
  h16* kh    = (h16*)alloc(P * 2);
  h16* vh    = (h16*)alloc(P * 2);
  h16* gh    = (h16*)alloc(P * 2);
  h16* oh    = (h16*)alloc(P * 2);
  h16* ogh   = (h16*)alloc(P * 2);
  float* xattn = (float*)alloc(P * 4);
  h16* xt    = (h16*)alloc(P * 2);
  h16* xtn   = (h16*)alloc(P * 2);
  h16* hk    = (h16*)alloc(P * 2);
  float* svbuf = (float*)alloc(P * 4);
  float* moe  = (float*)alloc(P * 4);
  float* sg   = (float*)alloc((size_t)NT * 4);
  float* msc  = (float*)alloc(16 * 4);
  float* moh  = (float*)alloc(16 * 4);
  int*   cnte = (int*)alloc(16 * 4);
  int*   ti   = (int*)alloc((size_t)NT * 2 * 4);
  float* twb  = (float*)alloc((size_t)NT * 2 * 4);
  int*   lists = (int*)alloc((size_t)Ee * NT * 4);
  h16* wqh = (h16*)alloc(W * 2);
  h16* wkh = (h16*)alloc(W * 2);
  h16* wvh = (h16*)alloc(W * 2);
  h16* wgh = (h16*)alloc(W * 2);
  h16* woh = (h16*)alloc(W * 2);
  h16* skh = (h16*)alloc((size_t)Cc * Ff * 2);
  h16* svh = (h16*)alloc((size_t)Ff * Cc * 2);
  h16* w1h = (h16*)alloc((size_t)Ee * Cc * Ff * 2);
  h16* w2h = (h16*)alloc((size_t)Ee * Ff * Cc * 2);

  k_init<<<dim3(4096), 256, 0, stream>>>(moe, msc, moh, cnte, P);

  k_convert<<<dim3(1024), 256, 0, stream>>>(Wq, wqh, W);
  k_convert<<<dim3(1024), 256, 0, stream>>>(Wk, wkh, W);
  k_convert<<<dim3(1024), 256, 0, stream>>>(Wv, wvh, W);
  k_convert<<<dim3(1024), 256, 0, stream>>>(Wg, wgh, W);
  k_convert<<<dim3(1024), 256, 0, stream>>>(Wo, woh, W);
  k_convert<<<dim3(1024), 256, 0, stream>>>(skw, skh, (size_t)Cc * Ff);
  k_convert<<<dim3(1024), 256, 0, stream>>>(svw, svh, (size_t)Ff * Cc);
  k_convert<<<dim3(4096), 256, 0, stream>>>(W1, w1h, (size_t)Ee * Cc * Ff);
  k_convert<<<dim3(4096), 256, 0, stream>>>(W2, w2h, (size_t)Ee * Ff * Cc);

  // attention half
  k_rmsnorm<float><<<dim3(NT), 256, 0, stream>>>(x, xn);
  dim3 gg(NT / 64, 16);
  k_gemm<0><<<gg, 128, 0, stream>>>(xn, wqh, nullptr, nullptr, nullptr, qh, NT, Hh * Nn, Cc);
  k_gemm<0><<<gg, 128, 0, stream>>>(xn, wkh, nullptr, nullptr, nullptr, kh, NT, Hh * Nn, Cc);
  k_gemm<0><<<gg, 128, 0, stream>>>(xn, wvh, nullptr, nullptr, nullptr, vh, NT, Hh * Nn, Cc);
  k_gemm<0><<<gg, 128, 0, stream>>>(xn, wgh, nullptr, nullptr, nullptr, gh, NT, Hh * Nn, Cc);
  k_rope<<<dim3(NT, 2), 256, 0, stream>>>(qh, cosb, sinb);
  k_rope<<<dim3(NT, 2), 256, 0, stream>>>(kh, cosb, sinb);
  k_attn<<<dim3(Tt / 64, Bb * Hh), 128, 0, stream>>>(qh, kh, vh, oh);
  k_og<<<dim3(2048), 256, 0, stream>>>(oh, gh, ogh, P);
  k_gemm<1><<<gg, 128, 0, stream>>>(ogh, woh, x, nullptr, xattn, nullptr, NT, Cc, Hh * Nn);

  // MoE half
  k_rmsnorm<float><<<dim3(NT), 256, 0, stream>>>(xattn, xt);
  k_router<<<dim3(NT / 4), 128, 0, stream>>>(xt, gate_w, ti, twb, msc, moh, cnte, lists);
  k_rmsnorm<h16><<<dim3(NT), 256, 0, stream>>>(xt, xtn);
  k_gemm<2><<<gg, 128, 0, stream>>>(xtn, skh, nullptr, skb, nullptr, hk, NT, Ff, Cc);
  k_gemm<3><<<gg, 128, 0, stream>>>(hk, svh, nullptr, nullptr, svbuf, nullptr, NT, Cc, Ff);
  k_sg<<<dim3(NT / 4), 128, 0, stream>>>(xt, sgw, sg);
  k_moe<<<dim3(NT / 16, Ee), 128, 0, stream>>>(xt, w1h, w2h, cnte, lists, twb, moe);

  k_combine<<<dim3(2048), 256, 0, stream>>>(xattn, moe, sg, svbuf, svb, (float*)d_out, P);
  k_aux<<<1, 32, 0, stream>>>(msc, moh, (float*)d_out + P);
}
